// Linear_26422638805106
// MI455X (gfx1250) — compile-verified, matrix operand order checked by reading, not run on hardware
//
#include <hip/hip_runtime.h>
#include <stdint.h>

// ---------------------------------------------------------------------------
// LoRA-augmented block-dequant linear for MI455X (gfx1250, wave32, WMMA).
//   y = x @ dequant(W)^T + 2.0 * (x @ Ld^T) @ Lu^T
// M=8192, N=4096, K=4096.  Prep pass -> bf16 working set (96 MB, L2-resident).
// GEMM: 128x128 WG tile, 8 waves of 32x64, K staged 32-deep through LDS with
// double buffering. Staging via Tensor Data Mover (TENSOR_LOAD_TO_LDS, with
// hardware LDS row padding), fallback GLOBAL_LOAD_ASYNC_TO_LDS_B128.
// LoRA fused as one extra zero-padded k-step.
// ---------------------------------------------------------------------------

#define M_DIM 8192
#define N_DIM 4096
#define K_DIM 4096
#define KT    32
#define LDSS  40                      // 32 + 8 pad (row stride 80B, 16B aligned)
#define ABUF_ELEMS (128 * LDSS)       // 5120 bf16 per matrix per buffer
#define BUF_ELEMS  (2 * ABUF_ELEMS)   // A + B per buffer
#define SMEM_ELEMS (2 * BUF_ELEMS)    // double buffered: 40960 B total

#define USE_TDM 1
#if __has_builtin(__builtin_amdgcn_global_load_async_to_lds_b128)
#define USE_ASYNC 1
#else
#define USE_ASYNC 0
#endif

typedef __bf16 bf16_t;
typedef bf16_t bf16x16 __attribute__((ext_vector_type(16)));
typedef float  f32x8   __attribute__((ext_vector_type(8)));
typedef float  f32x4   __attribute__((ext_vector_type(4)));
typedef unsigned int u32x4 __attribute__((ext_vector_type(4)));
typedef unsigned int u32x8 __attribute__((ext_vector_type(8)));
typedef unsigned int u32x2 __attribute__((ext_vector_type(2)));
typedef int          i32x4 __attribute__((ext_vector_type(4)));

// async-load builtin takes v4i32 pointers in global (as1) / LDS (as3) space
typedef __attribute__((address_space(1))) i32x4 gi32x4;
typedef __attribute__((address_space(3))) i32x4 li32x4;
typedef __attribute__((address_space(3))) unsigned short lds_ushort;

union FragU { bf16x16 v; u32x4 q[2]; };

__device__ __forceinline__ unsigned int f2bf(float f) {
  unsigned int u = __float_as_uint(f);
  u += 0x7FFFu + ((u >> 16) & 1u);          // round-to-nearest-even
  return u >> 16;
}

// ---- prep: x f32 -> bf16 (4 elems / thread) -------------------------------
__global__ void k_cvt_x(const float* __restrict__ x, unsigned short* __restrict__ xb) {
  int i = blockIdx.x * blockDim.x + threadIdx.x;     // float4 index
  f32x4 v = ((const f32x4*)x)[i];
  u32x2 o;
  o.x = f2bf(v.x) | (f2bf(v.y) << 16);
  o.y = f2bf(v.z) | (f2bf(v.w) << 16);
  ((u32x2*)xb)[i] = o;
}

// ---- prep: W * blockscale -> bf16 -----------------------------------------
__global__ void k_dequant_w(const float* __restrict__ w, const float* __restrict__ sc,
                            unsigned short* __restrict__ wb) {
  int i = blockIdx.x * blockDim.x + threadIdx.x;     // float4 index
  int col4 = i & (K_DIM / 4 - 1);
  int row  = i >> 10;                                 // K_DIM/4 == 1024
  float s = sc[(row >> 7) * (K_DIM / 128) + (col4 >> 5)];
  f32x4 v = ((const f32x4*)w)[i];
  u32x2 o;
  o.x = f2bf(v.x * s) | (f2bf(v.y * s) << 16);
  o.y = f2bf(v.z * s) | (f2bf(v.w * s) << 16);
  ((u32x2*)wb)[i] = o;
}

// ---- prep: t[row, j] = 2.0 * sum_k x[row,k]*Ld[j,k], bf16, K-padded to 32 --
__global__ void k_lora_t(const float* __restrict__ x, const float* __restrict__ ld,
                         unsigned short* __restrict__ tb) {
  int idx = blockIdx.x * blockDim.x + threadIdx.x;   // row*32 + j
  int j   = idx & 31;
  int row = idx >> 5;
  float acc = 0.0f;
  if (j < 16) {
    const float* xr = x  + (size_t)row * K_DIM;
    const float* lr = ld + (size_t)j   * K_DIM;
    #pragma unroll 4
    for (int k = 0; k < K_DIM; ++k) acc += xr[k] * lr[k];
    acc *= 2.0f;                                      // LORA_ALPHA / LORA_RANK
  }
  tb[idx] = (j < 16) ? (unsigned short)f2bf(acc) : (unsigned short)0;
}

// ---- prep: Lu [N,16] -> bf16 [N,32] zero-padded ---------------------------
__global__ void k_pad_up(const float* __restrict__ up, unsigned short* __restrict__ ub) {
  int idx = blockIdx.x * blockDim.x + threadIdx.x;   // o*32 + j
  int j = idx & 31;
  int o = idx >> 5;
  ub[idx] = (j < 16) ? (unsigned short)f2bf(up[o * 16 + j]) : (unsigned short)0;
}

#if USE_TDM
// ---- TDM: one descriptor moves a 128x32 bf16 tile (row stride K_DIM) into
//      LDS with hardware padding: 16 DWORDs data then 4 DWORDs pad = LDSS=40.
__device__ __forceinline__ void tdm_load_tile(unsigned long long gaddr, unsigned lds_byte) {
  u32x4 g0;
  g0[0] = 1u;                                            // count=1, user D#
  g0[1] = lds_byte;                                      // bits 63:32 lds_addr
  g0[2] = (unsigned)gaddr;                               // bits 95:64
  g0[3] = ((unsigned)(gaddr >> 32) & 0x01FFFFFFu)        // addr[56:32]
        | (2u << 30);                                    // type=2 (image)
  u32x8 g1;
  g1[0] = (1u << 16)      // data_size = 2 bytes
        | (1u << 20)      // pad_enable
        | (3u << 22)      // pad_interval: 16 DWORDs (64B) between pads
        | (3u << 25);     // pad_amount:   4 DWORDs (16B)
  g1[1] = ((unsigned)K_DIM & 0xFFFFu) << 16;             // tensor_dim0[15:0]
  g1[2] = (((unsigned)K_DIM >> 16) & 0xFFFFu)            // tensor_dim0[31:16]
        | (128u << 16);                                  // tensor_dim1[15:0]
  g1[3] = (32u << 16);                                   // tile_dim0 = 32 elems
  g1[4] = 128u;                                          // tile_dim1 = 128 rows
  g1[5] = (unsigned)K_DIM;                               // tensor_dim0_stride lo32
  g1[6] = 0u;                                            // stride0 hi / stride1 lo
  g1[7] = 0u;
  u32x4 gz = {0u, 0u, 0u, 0u};                           // groups 2/3 unused (2D)
  asm volatile("tensor_load_to_lds %0, %1, %2, %3"
               :: "s"(g0), "s"(g1), "s"(gz), "s"(gz) : "memory");
}
#endif

// ---- stage one 128x32 A-tile + 128x32 B-tile into LDS buffer `buf` --------
__device__ __forceinline__ void stage_tile(const unsigned short* __restrict__ xb,
                                           const unsigned short* __restrict__ wb,
                                           unsigned short* smem, unsigned lbase,
                                           int buf, int gm0, int gn0, int kt,
                                           int tid, int wave) {
  const unsigned aL = (unsigned)buf * BUF_ELEMS;
  const unsigned bL = aL + ABUF_ELEMS;
#if USE_TDM
  (void)tid;
  if (wave == 0) {
    tdm_load_tile((unsigned long long)(size_t)(xb + (size_t)gm0 * K_DIM + kt), lbase + 2u * aL);
    tdm_load_tile((unsigned long long)(size_t)(wb + (size_t)gn0 * K_DIM + kt), lbase + 2u * bL);
  }
#else
  (void)wave; (void)lbase;
  #pragma unroll
  for (int i = 0; i < 2; ++i) {
    int c    = tid + i * 256;           // chunk id in [0,512)
    int row  = c >> 2;
    int col8 = c & 3;
    unsigned gOffA = (unsigned)((gm0 + row) * K_DIM + kt + col8 * 8);
    unsigned gOffB = (unsigned)((gn0 + row) * K_DIM + kt + col8 * 8);
    unsigned lOff  = (unsigned)(row * LDSS + col8 * 8);
#if USE_ASYNC
    __builtin_amdgcn_global_load_async_to_lds_b128(
        (gi32x4*)(xb + gOffA), (li32x4*)(smem + aL + lOff), 0, 0);
    __builtin_amdgcn_global_load_async_to_lds_b128(
        (gi32x4*)(wb + gOffB), (li32x4*)(smem + bL + lOff), 0, 0);
#else
    *(u32x4*)(smem + aL + lOff) = *(const u32x4*)(xb + gOffA);
    *(u32x4*)(smem + bL + lOff) = *(const u32x4*)(wb + gOffB);
#endif
  }
#endif
}

__device__ __forceinline__ void wait_stage(int wave) {
#if USE_TDM
  if (wave == 0)
    asm volatile("s_wait_tensorcnt 0x0" ::: "memory");
#elif USE_ASYNC
  (void)wave;
#if __has_builtin(__builtin_amdgcn_s_wait_asynccnt)
  __builtin_amdgcn_s_wait_asynccnt(0);
#else
  asm volatile("s_wait_asynccnt 0x0" ::: "memory");
#endif
#else
  (void)wave;
#endif
}

// ---- main GEMM ------------------------------------------------------------
__launch_bounds__(256)
__global__ void k_gemm(const unsigned short* __restrict__ xb,
                       const unsigned short* __restrict__ wb,
                       const unsigned short* __restrict__ tb,
                       const unsigned short* __restrict__ ub,
                       float* __restrict__ out) {
  __shared__ __align__(16) unsigned short smem[SMEM_ELEMS];

  const int tid  = threadIdx.x;
  const int lane = tid & 31;
  const int wave = tid >> 5;
  const int wm   = wave >> 1;          // 0..3 -> 32-row slab
  const int wn   = wave & 1;           // 0..1 -> 64-col slab
  const int r    = lane & 15;          // M (A) / N (B) within a 16x16 tile
  const int hi   = lane >> 4;          // low/high K half per ISA layout

  const int gm0 = blockIdx.y * 128;
  const int gn0 = blockIdx.x * 128;

  // byte address of smem inside the LDS aperture (for TDM descriptors)
  const unsigned lbase = (unsigned)(size_t)(lds_ushort*)smem;

  f32x8 acc[2][4];
  #pragma unroll
  for (int i = 0; i < 2; ++i)
    #pragma unroll
    for (int j = 0; j < 4; ++j)
      #pragma unroll
      for (int e = 0; e < 8; ++e) acc[i][j][e] = 0.0f;

  // prologue: fill buffer 0
  stage_tile(xb, wb, smem, lbase, 0, gm0, gn0, 0, tid, wave);
  wait_stage(wave);
  __syncthreads();

  const unsigned aFragOff = (unsigned)((wm * 32 + r) * LDSS + hi * 8);
  const unsigned bFragOff = (unsigned)((wn * 64 + r) * LDSS + hi * 8);

  for (int kt = 0; kt < K_DIM; kt += KT) {
    const int cur = (kt >> 5) & 1;
    if (kt + KT < K_DIM)                       // prefetch next tile into other buf
      stage_tile(xb, wb, smem, lbase, 1 - cur, gm0, gn0, kt + KT, tid, wave);

    const unsigned short* A = smem + (unsigned)cur * BUF_ELEMS;
    const unsigned short* B = A + ABUF_ELEMS;

    FragU a[2], b[4];
    #pragma unroll
    for (int mt = 0; mt < 2; ++mt) {           // A frag: lane r = row, split K halves
      const unsigned short* p = A + aFragOff + mt * 16 * LDSS;
      a[mt].q[0] = *(const u32x4*)p;
      a[mt].q[1] = *(const u32x4*)(p + 16);
    }
    #pragma unroll
    for (int nt = 0; nt < 4; ++nt) {           // B frag from W rows (N-major, K contig)
      const unsigned short* p = B + bFragOff + nt * 16 * LDSS;
      b[nt].q[0] = *(const u32x4*)p;
      b[nt].q[1] = *(const u32x4*)(p + 16);
    }
    #pragma unroll
    for (int mt = 0; mt < 2; ++mt)
      #pragma unroll
      for (int nt = 0; nt < 4; ++nt)
        acc[mt][nt] = __builtin_amdgcn_wmma_f32_16x16x32_bf16(
            false, a[mt].v, false, b[nt].v, (short)0, acc[mt][nt], false, false);

    wait_stage(wave);                          // next-buffer DMA landed (wave 0)
    __syncthreads();                           // publish + everyone done with cur
  }

  // ---- LoRA epilogue: one extra k-step (K=16 zero-padded to 32) -----------
  {
    FragU a[2], b[4];
    #pragma unroll
    for (int mt = 0; mt < 2; ++mt) {
      const unsigned short* p = tb + (unsigned)((gm0 + wm * 32 + mt * 16 + r) * 32 + hi * 8);
      a[mt].q[0] = *(const u32x4*)p;
      a[mt].q[1] = *(const u32x4*)(p + 16);
    }
    #pragma unroll
    for (int nt = 0; nt < 4; ++nt) {
      const unsigned short* p = ub + (unsigned)((gn0 + wn * 64 + nt * 16 + r) * 32 + hi * 8);
      b[nt].q[0] = *(const u32x4*)p;
      b[nt].q[1] = *(const u32x4*)(p + 16);
    }
    #pragma unroll
    for (int mt = 0; mt < 2; ++mt)
      #pragma unroll
      for (int nt = 0; nt < 4; ++nt)
        acc[mt][nt] = __builtin_amdgcn_wmma_f32_16x16x32_bf16(
            false, a[mt].v, false, b[nt].v, (short)0, acc[mt][nt], false, false);
  }

  // ---- store: D layout lane r = N, VGPR j -> M = hi*8 + j -----------------
  #pragma unroll
  for (int mt = 0; mt < 2; ++mt) {
    int rowBase = gm0 + wm * 32 + mt * 16 + hi * 8;
    #pragma unroll
    for (int nt = 0; nt < 4; ++nt) {
      unsigned colOff = (unsigned)(gn0 + wn * 64 + nt * 16 + r);
      #pragma unroll
      for (int j = 0; j < 8; ++j)
        out[(unsigned)(rowBase + j) * N_DIM + colOff] = acc[mt][nt][j];
    }
  }
}

// ---------------------------------------------------------------------------
extern "C" void kernel_launch(void* const* d_in, const int* in_sizes, int n_in,
                              void* d_out, int out_size, void* d_ws, size_t ws_size,
                              hipStream_t stream) {
  (void)in_sizes; (void)n_in; (void)out_size; (void)ws_size;
  const float* x  = (const float*)d_in[0];   // [8192, 4096]
  const float* w  = (const float*)d_in[1];   // [4096, 4096]
  const float* sc = (const float*)d_in[2];   // [32, 32]
  const float* ld = (const float*)d_in[3];   // [16, 4096]
  const float* up = (const float*)d_in[4];   // [4096, 16]
  float* out = (float*)d_out;                // [8192, 4096]

  char* ws = (char*)d_ws;
  unsigned short* xb = (unsigned short*)(ws);                            // 64 MB
  unsigned short* wb = (unsigned short*)(ws + (size_t)64 * 1024 * 1024); // 32 MB
  unsigned short* tb = (unsigned short*)(ws + (size_t)96 * 1024 * 1024); // 512 KB
  unsigned short* ub = (unsigned short*)(ws + (size_t)96 * 1024 * 1024 + 524288);

  k_cvt_x    <<<(M_DIM * K_DIM / 4) / 256, 256, 0, stream>>>(x, xb);
  k_dequant_w<<<(N_DIM * K_DIM / 4) / 256, 256, 0, stream>>>(w, sc, wb);
  k_lora_t   <<<(M_DIM * 32) / 256,        256, 0, stream>>>(x, ld, tb);
  k_pad_up   <<<(N_DIM * 32) / 256,        256, 0, stream>>>(up, ub);

  dim3 grid(N_DIM / 128, M_DIM / 128);     // 32 x 64 workgroups
  k_gemm<<<grid, 256, 0, stream>>>(xb, wb, tb, ub, out);
}
